// Retention_69879117906727
// MI455X (gfx1250) — matrix operand order repaired and checked by reading, not compiled, not run on hardware
//
#include <hip/hip_runtime.h>
#include <hip/hip_bf16.h>
#include <cstdint>

typedef __attribute__((ext_vector_type(16))) __bf16 v16bf;
typedef __attribute__((ext_vector_type(8)))  float  v8f;
typedef __attribute__((ext_vector_type(4)))  float  v4f;

#define SEQ     4096
#define NBAT    8
#define DMODEL  256
// gamma = 0.96875
#define LN_GAMMA (-0.031748698315134226f)

// LDS row strides (dwords), padded to avoid bank conflicts
#define QK_STRIDE 130   // 64-row x 128-dword bf16 tiles (Q tile, K tiles)
#define V_STRIDE  258   // 32-row x 256-dword pair-packed V tiles
#define S_STRIDE  34    // 64-row x 32-dword bf16 S tile

// LDS layout offsets (dwords)
#define QLDS_OFF  0
#define KT_OFF    (64 * QK_STRIDE)               // two K buffers follow Q
#define KT_SIZE   (64 * QK_STRIDE)
#define VT_OFF    (3 * 64 * QK_STRIDE)
#define VT_SIZE   (32 * V_STRIDE)
#define S32_OFF   (VT_OFF + 2 * VT_SIZE)
#define SMEM_DW   (S32_OFF + 64 * S_STRIDE)

__device__ __forceinline__ uint16_t f2bf(float f) {
  uint32_t u = __float_as_uint(f);
  u += 0x7fffu + ((u >> 16) & 1u);
  return (uint16_t)(u >> 16);
}
__device__ __forceinline__ uint32_t pack2bf(float a, float b) {
  return (uint32_t)f2bf(a) | ((uint32_t)f2bf(b) << 16);
}

union Frag { uint32_t u[8]; v16bf v; };

// ---- CDNA5 async global->LDS copy (ASYNCcnt path) -------------------------
__device__ __forceinline__ uint32_t lds_off(const void* p) {
  return (uint32_t)(uintptr_t)p;          // low 32 bits of generic addr = LDS offset
}
__device__ __forceinline__ void async_b128(uint32_t lds_byte_off, const void* g) {
  asm volatile("global_load_async_to_lds_b128 %0, %1, off"
               :: "v"(lds_byte_off), "v"((uint64_t)(uintptr_t)g) : "memory");
}
__device__ __forceinline__ void wait_async0() {
  asm volatile("s_wait_asynccnt 0x0" ::: "memory");
}

// ---------------------------------------------------------------------------
// Kernel 0: pack W_Q/W_K/W_V (256x256 f32 [k][n]) into bf16 with k-pairs
// adjacent: Wp_dword[(k>>1)*256 + n] = {bf16(W[k][n]), bf16(W[k+1][n])}.
// ---------------------------------------------------------------------------
__global__ __launch_bounds__(256) void pack_w_kernel(
    const float* __restrict__ Wq, const float* __restrict__ Wk,
    const float* __restrict__ Wv, uint32_t* __restrict__ Wp)
{
  int id = blockIdx.x * 256 + threadIdx.x;     // 0 .. 3*128*256
  int w  = id / (128 * 256);
  int r  = id % (128 * 256);
  int kk = r / 256;
  int n  = r % 256;
  const float* W = (w == 0) ? Wq : (w == 1) ? Wk : Wv;
  float a = W[(2 * kk) * 256 + n];
  float b = W[(2 * kk + 1) * 256 + n];
  Wp[w * (128 * 256) + kk * 256 + n] = pack2bf(a, b);
}

// ---------------------------------------------------------------------------
// Kernel 1: zero the strictly-upper (m > n) region of ret (causal mask).
// Non-temporal float4 stores; scalar fallback only on the diagonal straddle.
// ---------------------------------------------------------------------------
__global__ __launch_bounds__(256) void fill_upper_kernel(float* __restrict__ ret)
{
  size_t id = ((size_t)blockIdx.x * 256 + threadIdx.x) * 4;
  int m = (int)(id % SEQ);
  int n = (int)((id / SEQ) % SEQ);
  if (m > n) {
    v4f z = {0.f, 0.f, 0.f, 0.f};
    __builtin_nontemporal_store(z, (v4f*)(ret + id));
  } else if (m + 3 > n) {
#pragma unroll
    for (int t = 0; t < 4; ++t)
      if (m + t > n) __builtin_nontemporal_store(0.f, ret + id + t);
  }
}

// ---------------------------------------------------------------------------
// Kernel 2: projections + xPos RoPE (bf16 WMMA, f32 accumulate).
// ---------------------------------------------------------------------------
__global__ __launch_bounds__(256) void proj_rope_kernel(
    const float* __restrict__ x, const uint32_t* __restrict__ Wp,
    uint16_t* __restrict__ Qbf, uint16_t* __restrict__ Kbf,
    uint32_t* __restrict__ Vp)
{
  __shared__ uint32_t xlds[64 * QK_STRIDE];
  const int b = blockIdx.y, rb = blockIdx.x;
  const int tid = threadIdx.x, wave = tid >> 5, lane = tid & 31;
  const int half = lane >> 4, lm = lane & 15;
  const int irow = wave >> 1;
  const int jbase = (wave & 1) * 8;

  const float* xt = x + ((size_t)(b * SEQ + rb * 64)) * DMODEL;
#pragma unroll 4
  for (int i = tid; i < 64 * 128; i += 256) {
    int row = i >> 7, c = i & 127;
    xlds[row * QK_STRIDE + c] = pack2bf(xt[2 * i], xt[2 * i + 1]);
  }
  __syncthreads();

  for (int w = 0; w < 3; ++w) {
    const uint32_t* W = Wp + w * (128 * 256);
    v8f acc[8];
#pragma unroll
    for (int t = 0; t < 8; ++t)
#pragma unroll
      for (int e = 0; e < 8; ++e) acc[t][e] = 0.f;

#pragma unroll 1
    for (int kb = 0; kb < 8; ++kb) {
      Frag A;
#pragma unroll
      for (int v2 = 0; v2 < 8; ++v2) {
        int doff = kb * 32 + half * 16 + v2 * 2;
        A.u[v2] = xlds[(irow * 16 + lm) * QK_STRIDE + (doff >> 1)];
      }
#pragma unroll
      for (int t = 0; t < 8; ++t) {
        int col = (jbase + t) * 16 + lm;
        Frag Bf;
#pragma unroll
        for (int v2 = 0; v2 < 8; ++v2)
          Bf.u[v2] = W[(kb * 16 + half * 8 + v2) * 256 + col];
        acc[t] = __builtin_amdgcn_wmma_f32_16x16x32_bf16(
            false, A.v, false, Bf.v, (short)0, acc[t], false, false);
      }
    }

    if (w == 2) {
#pragma unroll
      for (int t = 0; t < 8; ++t) {
        int col = (jbase + t) * 16 + lm;
#pragma unroll
        for (int v2 = 0; v2 < 8; v2 += 2) {
          int mrow = rb * 64 + irow * 16 + half * 8 + v2;   // even
          Vp[((size_t)(b * (SEQ / 2)) + (mrow >> 1)) * 256 + col] =
              pack2bf(acc[t][v2], acc[t][v2 + 1]);
        }
      }
    } else {
      uint16_t* dst = (w == 0) ? Qbf : Kbf;
#pragma unroll
      for (int t = 0; t < 8; ++t) {
        int col = (jbase + t) * 16 + lm;
        int tp = col >> 1;
        float inv_freq = __powf(10000.f, -(float)(2 * tp) / (float)DMODEL);
        float basev = ((float)(2 * tp) + 0.4f * DMODEL) / (1.4f * DMODEL);
        float lbase = __log2f(basev);
#pragma unroll
        for (int v2 = 0; v2 < 8; ++v2) {
          int n = rb * 64 + irow * 16 + half * 8 + v2;
          float val = acc[t][v2];
          float p = __shfl_xor(val, 1, 32);
          float s, c;
          __sincosf((float)n * inv_freq, &s, &c);
          float r = (col & 1) ? (p * s + val * c) : (val * c - p * s);
          float scl = exp2f(lbase * ((float)n / 512.f));
          if (w == 1) scl = 1.f / scl;
          dst[((size_t)(b * SEQ + n)) * DMODEL + col] = f2bf(r * scl);
        }
      }
    }
  }
}

// ---------------------------------------------------------------------------
// Kernel 3: fused causal retention, async double-buffered K/V tiles in LDS.
// ---------------------------------------------------------------------------
__global__ __launch_bounds__(256) void retention_kernel(
    const uint16_t* __restrict__ Qbf, const uint16_t* __restrict__ Kbf,
    const uint32_t* __restrict__ Vp, float* __restrict__ out,
    float* __restrict__ ret)
{
  extern __shared__ uint32_t smem[];
  uint32_t* qlds = smem + QLDS_OFF;
  uint32_t* s32  = smem + S32_OFF;
  uint16_t* s16  = (uint16_t*)s32;
  const uint32_t smem_base = lds_off(smem);

  const int b = blockIdx.y, nb = blockIdx.x;
  const int tid = threadIdx.x, wave = tid >> 5, lane = tid & 31;
  const int half = lane >> 4, lm = lane & 15;
  const int irow = wave >> 1;
  const int jbase = (wave & 1) * 8;
  const int sjbase = (wave & 1) * 2;

  // ---- async stage of a K/V block pair -----------------------------------
  auto stage_kv = [&](int bufi, int mb2) {
    uint32_t koff = smem_base + (uint32_t)(KT_OFF + bufi * KT_SIZE) * 4u;
    const char* gk = (const char*)Kbf + ((size_t)(b * SEQ + mb2 * 64)) * 512;
#pragma unroll
    for (int it = 0; it < 8; ++it) {
      int c0 = tid + it * 256;
      int row = c0 >> 5, ch = (c0 & 31) * 4;   // chunk start (dwords)
      async_b128(koff + (uint32_t)(row * QK_STRIDE + ch) * 4u,
                 gk + (size_t)row * 512 + (size_t)ch * 4);
    }
    uint32_t voff = smem_base + (uint32_t)(VT_OFF + bufi * VT_SIZE) * 4u;
    const char* gv = (const char*)Vp +
        ((size_t)(b * (SEQ / 2)) + (size_t)mb2 * 32) * 1024;
#pragma unroll
    for (int it = 0; it < 8; ++it) {
      int c0 = tid + it * 256;
      int row = c0 >> 6, ch = (c0 & 63) * 4;
      async_b128(voff + (uint32_t)(row * V_STRIDE + ch) * 4u,
                 gv + (size_t)row * 1024 + (size_t)ch * 4);
    }
  };

  // ---- Q tile + first K/V block via async DMA ----------------------------
  {
    uint32_t qoff = smem_base + (uint32_t)QLDS_OFF * 4u;
    const char* gq = (const char*)Qbf + ((size_t)(b * SEQ + nb * 64)) * 512;
#pragma unroll
    for (int it = 0; it < 8; ++it) {
      int c0 = tid + it * 256;
      int row = c0 >> 5, ch = (c0 & 31) * 4;
      async_b128(qoff + (uint32_t)(row * QK_STRIDE + ch) * 4u,
                 gq + (size_t)row * 512 + (size_t)ch * 4);
    }
  }
  stage_kv(0, 0);
  wait_async0();
  __syncthreads();

  v8f oacc[8];
#pragma unroll
  for (int t = 0; t < 8; ++t)
#pragma unroll
    for (int e = 0; e < 8; ++e) oacc[t][e] = 0.f;

  int buf = 0;
  for (int mb = 0; mb <= nb; ++mb) {
    if (mb < nb) stage_kv(buf ^ 1, mb + 1);    // prefetch next block

    const uint32_t* kt = smem + KT_OFF + buf * KT_SIZE;
    const uint32_t* vt = smem + VT_OFF + buf * VT_SIZE;

    // ---- S = Q . K^T (LDS-fed WMMA) --------------------------------------
    v8f sacc[2];
#pragma unroll
    for (int u = 0; u < 2; ++u)
#pragma unroll
      for (int e = 0; e < 8; ++e) sacc[u][e] = 0.f;

#pragma unroll 1
    for (int kb = 0; kb < 8; ++kb) {
      Frag A;
#pragma unroll
      for (int v2 = 0; v2 < 8; ++v2) {
        int doff = kb * 32 + half * 16 + v2 * 2;
        A.u[v2] = qlds[(irow * 16 + lm) * QK_STRIDE + (doff >> 1)];
      }
#pragma unroll
      for (int u = 0; u < 2; ++u) {
        int mloc = (sjbase + u) * 16 + lm;
        Frag Bf;
#pragma unroll
        for (int v2 = 0; v2 < 8; ++v2)
          Bf.u[v2] = kt[mloc * QK_STRIDE + kb * 16 + half * 8 + v2];
        sacc[u] = __builtin_amdgcn_wmma_f32_16x16x32_bf16(
            false, A.v, false, Bf.v, (short)0, sacc[u], false, false);
      }
    }

    // ---- decay mask, NT ret write, stage bf16 S in LDS -------------------
#pragma unroll
    for (int u = 0; u < 2; ++u) {
      int mloc = (sjbase + u) * 16 + lm;
      int mg = mb * 64 + mloc;
#pragma unroll
      for (int v2 = 0; v2 < 8; ++v2) {
        int nloc = irow * 16 + half * 8 + v2;
        int ng = nb * 64 + nloc;
        float d = (ng >= mg) ? __expf((float)(ng - mg) * LN_GAMMA) : 0.f;
        float val = sacc[u][v2] * d;
        __builtin_nontemporal_store(val, ret + ((size_t)b * SEQ + ng) * SEQ + mg);
        s16[nloc * (2 * S_STRIDE) + mloc] = f2bf(val);
      }
    }
    __syncthreads();

    // ---- out += S . V ----------------------------------------------------
#pragma unroll
    for (int kb = 0; kb < 2; ++kb) {
      Frag A;
#pragma unroll
      for (int v2 = 0; v2 < 8; ++v2) {
        int moff = kb * 32 + half * 16 + v2 * 2;
        A.u[v2] = s32[(irow * 16 + lm) * S_STRIDE + (moff >> 1)];
      }
#pragma unroll
      for (int t = 0; t < 8; ++t) {
        int col = (jbase + t) * 16 + lm;
        Frag Bf;
#pragma unroll
        for (int v2 = 0; v2 < 8; ++v2)
          Bf.u[v2] = vt[(kb * 16 + half * 8 + v2) * V_STRIDE + col];
        oacc[t] = __builtin_amdgcn_wmma_f32_16x16x32_bf16(
            false, A.v, false, Bf.v, (short)0, oacc[t], false, false);
      }
    }

    wait_async0();       // next K/V buffer resident
    __syncthreads();     // all reads of s16 / old buffers done
    buf ^= 1;
  }

  // ---- store out (streaming, NT) -----------------------------------------
#pragma unroll
  for (int t = 0; t < 8; ++t) {
    int col = (jbase + t) * 16 + lm;
#pragma unroll
    for (int v2 = 0; v2 < 8; ++v2) {
      int ng = nb * 64 + irow * 16 + half * 8 + v2;
      __builtin_nontemporal_store(oacc[t][v2],
                                  out + ((size_t)(b * SEQ + ng)) * DMODEL + col);
    }
  }
}

// ---------------------------------------------------------------------------
extern "C" void kernel_launch(void* const* d_in, const int* in_sizes, int n_in,
                              void* d_out, int out_size, void* d_ws, size_t ws_size,
                              hipStream_t stream)
{
  (void)in_sizes; (void)n_in; (void)out_size; (void)ws_size;
  const float* x  = (const float*)d_in[0];
  const float* Wq = (const float*)d_in[1];
  const float* Wk = (const float*)d_in[2];
  const float* Wv = (const float*)d_in[3];

  float* out = (float*)d_out;                         // B*S*256 f32
  float* ret = out + (size_t)NBAT * SEQ * DMODEL;     // then B*S*S f32

  char* ws = (char*)d_ws;
  uint32_t* Wp  = (uint32_t*)ws;  ws += (size_t)3 * 128 * 256 * sizeof(uint32_t);
  uint16_t* Qbf = (uint16_t*)ws;  ws += (size_t)NBAT * SEQ * DMODEL * sizeof(uint16_t);
  uint16_t* Kbf = (uint16_t*)ws;  ws += (size_t)NBAT * SEQ * DMODEL * sizeof(uint16_t);
  uint32_t* Vp  = (uint32_t*)ws;  // B*(S/2)*256 dwords

  pack_w_kernel<<<3 * 128, 256, 0, stream>>>(Wq, Wk, Wv, Wp);

  dim3 grid(SEQ / 64, NBAT);
  proj_rope_kernel<<<grid, 256, 0, stream>>>(x, Wp, Qbf, Kbf, Vp);

  size_t relem = (size_t)NBAT * SEQ * SEQ;
  fill_upper_kernel<<<(uint32_t)(relem / 4 / 256), 256, 0, stream>>>(ret);

  retention_kernel<<<grid, 256, SMEM_DW * sizeof(uint32_t), stream>>>(
      Qbf, Kbf, Vp, out, ret);
}
